// MultiScaleRetinexNet_62088047231053
// MI455X (gfx1250) — matrix operand order repaired and checked by reading, not compiled, not run on hardware
//
#include <hip/hip_runtime.h>
#include <hip/hip_bf16.h>
#include <math.h>

// ---------------------------------------------------------------------------
// MultiScaleRetinexNet forward for gfx1250 (MI455X, wave32, WMMA).
// Every pointwise (1x1) conv / SSM projection runs through a templated WMMA
// GEMM (v_wmma_f32_16x16x32_f16, f16 in / f32 accumulate). Template params
// <C, LDW, O, ACT, HB, HB2> make the whole kernel branch-free: K padding and
// row guards are compile-time or clamp+cndmask, bias/activation are constexpr.
// ---------------------------------------------------------------------------

typedef __attribute__((ext_vector_type(16))) _Float16 v16h;
typedef __attribute__((ext_vector_type(8)))  float    v8f;

#define ACT_NONE 0
#define ACT_SILU 1
#define ACT_SIG  2
#define ACT_SP   3

template<int ACT>
__device__ __forceinline__ float actf(float v) {
  if (ACT == ACT_SILU) return v / (1.f + __expf(-v));
  if (ACT == ACT_SIG)  return 1.f / (1.f + __expf(-v));
  if (ACT == ACT_SP)   return (v > 20.f) ? v : log1pf(__expf(v));
  return v;
}

// ---------------- WMMA pointwise conv:
//   Y[b,o,n] = act(sum_c W[o,c] X[b,c,n] + bias[o] + bias2d[b,o])
// One wave computes a 16(M) x 32(N) tile (two WMMA accumulators sharing A).
// K padded to 32 with zeros via clamped unconditional loads + selects.
template<int C, int LDW, int O, int ACT, bool HB, bool HB2>
__global__ void k_pw_wmma(const float* __restrict__ X, const float* __restrict__ Wm,
                          const float* __restrict__ bias, const float* __restrict__ bias2d,
                          float* __restrict__ Y, int HW)
{
  constexpr bool MG = (O % 16) != 0;       // row guard needed only for O=1,3,8...
  const int lane = threadIdx.x;            // 0..31
  const int half = lane >> 4;
  const int l16  = lane & 15;
  const int Ntot = 8 * HW;
  const int ntile = blockIdx.x * blockDim.y + threadIdx.y;
  const int n0 = ntile * 32;
  if (n0 >= Ntot) return;                  // wave-uniform exit -> EXEC all ones at WMMA
  const int mtile = blockIdx.y;
  const int b   = n0 / HW;                 // HW % 32 == 0: tiles never straddle batch
  const int hw0 = n0 % HW;
  const float* Xb = X + b * C * HW + hw0 + l16;
  const int m  = mtile * 16 + l16;         // A row (lanes 0..15 / 16..31 same rows)
  const int mc = (MG && m >= O) ? 0 : m;   // clamped, always-valid row
  const float* Wrow = Wm + mc * LDW;

  v8f acc0 = {}, acc1 = {};
#pragma unroll
  for (int k0 = 0; k0 < C; k0 += 32) {
    v16h a, b0, b1;
#pragma unroll
    for (int i = 0; i < 16; ++i) {
      // A: ISA 16-bit 16x32 layout (lanes<16: K 0-7,16-23; lanes>=16: K 8-15,24-31)
      int kA  = half ? ((i < 8) ? k0 + i + 8 : k0 + i + 16)
                     : ((i < 8) ? k0 + i     : k0 + i + 8);
      int kAc = (kA < C) ? kA : 0;
      float av = Wrow[kAc];                            // unconditional load
      bool okA = (!MG || m < O) && (kA < C);
      av = okA ? av : 0.f;                             // cndmask, no branch
      a[i] = (_Float16)av;
      // B: same K order as A (lanes<16: K 0-7,16-23; lanes>=16: K 8-15,24-31)
      int kB  = k0 + half * 8 + i + ((i < 8) ? 0 : 8);
      int kBc = (kB < C) ? kB : 0;
      float bv0 = Xb[kBc * HW];                        // unconditional loads
      float bv1 = Xb[kBc * HW + 16];
      bool okB = (kB < C);
      b0[i] = (_Float16)(okB ? bv0 : 0.f);
      b1[i] = (_Float16)(okB ? bv1 : 0.f);
    }
    acc0 = __builtin_amdgcn_wmma_f32_16x16x32_f16(false, a, false, b0,
                                                  (short)0, acc0, false, false);
    acc1 = __builtin_amdgcn_wmma_f32_16x16x32_f16(false, a, false, b1,
                                                  (short)0, acc1, false, false);
  }
#pragma unroll
  for (int r = 0; r < 8; ++r) {            // C/D: VGPR r -> M=r (lanes<16) / r+8 (lanes>=16)
    int mo = mtile * 16 + r + (half ? 8 : 0);
    bool st = !MG || (mo < O);
    int moc = st ? mo : 0;                 // clamped indices: loads stay unconditional
    float extra = 0.f;
    if (HB)  extra += bias[moc];
    if (HB2) extra += bias2d[b * O + moc];
    float v0 = actf<ACT>(acc0[r] + extra);
    float v1 = actf<ACT>(acc1[r] + extra);
    if (st) {
      int base = (b * O + mo) * HW + hw0 + l16;
      Y[base]      = v0;
      Y[base + 16] = v1;
    }
  }
}

template<int C, int LDW, int O, int ACT, bool HB, bool HB2>
static void launch_pw(const float* X, const float* Wm, const float* bias, const float* b2d,
                      float* Y, int HW, hipStream_t stream)
{
  int Ntot = 8 * HW;
  dim3 bdim(32, 4), gdim((unsigned)((Ntot / 32 + 3) / 4), (unsigned)((O + 15) / 16));
  k_pw_wmma<C, LDW, O, ACT, HB, HB2><<<gdim, bdim, 0, stream>>>(X, Wm, bias, b2d, Y, HW);
}

// ---------------- depthwise 3x3, replicate padding, stride s
__global__ void k_dw3x3(const float* __restrict__ x, const float* __restrict__ w,
                        float* __restrict__ y, int C, int H, int Wd, int stride)
{
  int Ho = H / stride, Wo = Wd / stride;
  int total = 8 * C * Ho * Wo;
  int i = blockIdx.x * blockDim.x + threadIdx.x;
  if (i >= total) return;
  int wo = i % Wo; int t = i / Wo;
  int ho = t % Ho; t /= Ho;
  int c  = t % C;  int b = t / C;
  const float* xp = x + (b * C + c) * H * Wd;
  const float* wp = w + c * 9;
  float acc = 0.f;
#pragma unroll
  for (int ky = 0; ky < 3; ++ky) {
    int yy = ho * stride + ky - 1; yy = min(max(yy, 0), H - 1);
#pragma unroll
    for (int kx = 0; kx < 3; ++kx) {
      int xx = wo * stride + kx - 1; xx = min(max(xx, 0), Wd - 1);
      acc += wp[ky * 3 + kx] * xp[yy * Wd + xx];
    }
  }
  y[i] = acc;
}

// ---------------- GroupNorm(G) + affine + SiLU, in place. One block per (b, group).
__global__ void k_gn_silu(float* __restrict__ x, const float* __restrict__ g,
                          const float* __restrict__ bta, int C, int HW, int G)
{
  int bg = blockIdx.x, gr = bg % G, b = bg / G;
  int Cpg = C / G;
  int N = Cpg * HW;
  float* xp = x + (b * C + gr * Cpg) * HW;
  __shared__ float s1[256], s2[256];
  float a = 0.f, q = 0.f;
  for (int i = threadIdx.x; i < N; i += 256) { float v = xp[i]; a += v; q += v * v; }
  s1[threadIdx.x] = a; s2[threadIdx.x] = q; __syncthreads();
  for (int s = 128; s > 0; s >>= 1) {
    if (threadIdx.x < s) { s1[threadIdx.x] += s1[threadIdx.x + s]; s2[threadIdx.x] += s2[threadIdx.x + s]; }
    __syncthreads();
  }
  float m = s1[0] / (float)N, var = s2[0] / (float)N - m * m;
  float rs = rsqrtf(var + 1e-5f);
  for (int i = threadIdx.x; i < N; i += 256) {
    int c = gr * Cpg + i / HW;
    float v = (xp[i] - m) * rs * g[c] + bta[c];
    xp[i] = v / (1.f + __expf(-v));
  }
}

// ---------------- LayerNorm over channels at each (b,t); x,y: [B,C,HW]
__global__ void k_ln_chan(const float* __restrict__ x, const float* __restrict__ g,
                          const float* __restrict__ bta, float* __restrict__ y,
                          int C, int HW)
{
  int total = 8 * HW;
  int i = blockIdx.x * blockDim.x + threadIdx.x;
  if (i >= total) return;
  int b = i / HW, t = i % HW;
  const float* xp = x + b * C * HW + t;
  float a = 0.f, q = 0.f;
  for (int c = 0; c < C; ++c) { float v = xp[c * HW]; a += v; q += v * v; }
  float m = a / C, var = q / C - m * m, rs = rsqrtf(var + 1e-5f);
  float* yp = y + b * C * HW + t;
  for (int c = 0; c < C; ++c) yp[c * HW] = (xp[c * HW] - m) * rs * g[c] + bta[c];
}

// ---------------- FiLM constant term: dw of a constant field == const * sum9(kernel).
// femb[b,o] = sum_{c<32} pw48[o,16+c] * emb[b,c] * sum9(dw48[16+c])
__global__ void k_femb(const float* __restrict__ emb, const float* __restrict__ pw48,
                       const float* __restrict__ dw48, float* __restrict__ femb)
{
  int t = blockIdx.x * blockDim.x + threadIdx.x;
  if (t >= 128) return;
  int b = t >> 4, o = t & 15;
  float s = 0.f;
  for (int c = 0; c < 32; ++c) {
    float s9 = 0.f;
    for (int q = 0; q < 9; ++q) s9 += dw48[(16 + c) * 9 + q];
    s += pw48[o * 48 + 16 + c] * emb[b * 32 + c] * s9;
  }
  femb[t] = s;
}

// ---------------- FiLM apply: out = (1+gamma)*x + beta, mod=[B,32,HW]
__global__ void k_film_apply(const float* __restrict__ e0, const float* __restrict__ mod,
                             float* __restrict__ out, int HW)
{
  int total = 8 * 16 * HW;
  int i = blockIdx.x * blockDim.x + threadIdx.x;
  if (i >= total) return;
  int hw = i % HW; int r = i / HW;
  int o = r % 16, b = r / 16;
  const float* mb = mod + b * 32 * HW;
  float gamma = mb[o * HW + hw];
  float beta  = mb[(16 + o) * HW + hw];
  out[i] = (1.f + gamma) * e0[i] + beta;
}

// ---------------- SSM scan: one block per (batch, direction).
// h_t[i] = sum_j exp(dt_i*A_ij) h_{t-1}[j] + dt_i*Bm_i*xp_i
__global__ void k_ssm_scan(const float* __restrict__ dt, const float* __restrict__ xp,
                           const float* __restrict__ A, const float* __restrict__ Bm,
                           float* __restrict__ hs, int T)
{
  int b = blockIdx.x, dir = blockIdx.y, nb = gridDim.x;
  int tid = threadIdx.x, i = tid >> 4, j = tid & 15;
  const float* dtb = dt + b * 16 * T;
  const float* xpb = xp + b * 16 * T;
  float* hsb = hs + (dir * nb + b) * 16 * T;
  float Aij = A[i * 16 + j], Bmi = Bm[i];
  __shared__ float hsm[16];
  __shared__ float red[16][17];
  if (tid < 16) hsm[tid] = 0.f;
  __syncthreads();
  for (int t = 0; t < T; ++t) {
    int tt = dir ? (T - 1 - t) : t;
    float dti = dtb[i * T + tt];
    red[i][j] = __expf(dti * Aij) * hsm[j];
    __syncthreads();
    if (j == 0) {
      float s = 0.f;
#pragma unroll
      for (int q = 0; q < 16; ++q) s += red[i][q];
      s += dti * Bmi * xpb[i * T + tt];
      hsm[i] = s;
      hsb[i * T + tt] = s;
    }
    __syncthreads();
  }
}

// ---------------- misc elementwise / reshaping kernels
__global__ void k_avg(float* __restrict__ d, const float* __restrict__ a,
                      const float* __restrict__ b, int n)
{
  int i = blockIdx.x * blockDim.x + threadIdx.x;
  if (i < n) d[i] = 0.5f * (a[i] + b[i]);
}
__global__ void k_add(float* __restrict__ a, const float* __restrict__ b, int n)
{
  int i = blockIdx.x * blockDim.x + threadIdx.x;
  if (i < n) a[i] += b[i];
}
__global__ void k_yfix(float* __restrict__ y, const float* __restrict__ ln,
                       const float* __restrict__ D, int C, int HW, int n)
{
  int i = blockIdx.x * blockDim.x + threadIdx.x;
  if (i >= n) return;
  int d = (i / HW) % C;
  y[i] += D[d] * ln[i];
}
__global__ void k_copy_chan(float* __restrict__ dst, const float* __restrict__ src,
                            int Cd, int c0, int Cs, int HW)
{
  int total = 8 * Cs * HW;
  int i = blockIdx.x * blockDim.x + threadIdx.x;
  if (i >= total) return;
  int hw = i % HW; int r = i / HW;
  int c = r % Cs, b = r / Cs;
  dst[(b * Cd + c0 + c) * HW + hw] = src[i];
}
__global__ void k_pixshuf(const float* __restrict__ src, float* __restrict__ dst,
                          int C, int h, int w)   // src [8,4C,h,w] -> dst [8,C,2h,2w]
{
  int total = 8 * C * 4 * h * w;
  int i = blockIdx.x * blockDim.x + threadIdx.x;
  if (i >= total) return;
  int W2 = 2 * w;
  int x2 = i % W2; int r = i / W2;
  int y2 = r % (2 * h); r /= (2 * h);
  int c = r % C, b = r / C;
  int sc = c * 4 + (y2 & 1) * 2 + (x2 & 1);
  dst[i] = src[((b * 4 * C + sc) * h + (y2 >> 1)) * w + (x2 >> 1)];
}
__global__ void k_gap(const float* __restrict__ x, float* __restrict__ gap, int HW)
{
  int bc = blockIdx.x;
  const float* p = x + bc * HW;
  __shared__ float s1[256];
  float a = 0.f;
  for (int i = threadIdx.x; i < HW; i += 256) a += p[i];
  s1[threadIdx.x] = a; __syncthreads();
  for (int s = 128; s > 0; s >>= 1) {
    if (threadIdx.x < s) s1[threadIdx.x] += s1[threadIdx.x + s];
    __syncthreads();
  }
  if (threadIdx.x == 0) gap[bc] = s1[0] / (float)HW;
}
__global__ void k_eca_gate(const float* __restrict__ gap, const float* __restrict__ k,
                           float* __restrict__ gate)
{
  int t = blockIdx.x * blockDim.x + threadIdx.x;
  if (t >= 512) return;
  int c = t & 63;
  float l = (c > 0)  ? gap[t - 1] : 0.f;
  float m = gap[t];
  float rr = (c < 63) ? gap[t + 1] : 0.f;
  float v = k[0] * l + k[1] * m + k[2] * rr;
  gate[t] = 1.f / (1.f + __expf(-v));
}
__global__ void k_scale_chan(float* __restrict__ x, const float* __restrict__ gate,
                             int HW, int n)
{
  int i = blockIdx.x * blockDim.x + threadIdx.x;
  if (i >= n) return;
  x[i] *= gate[i / HW];
}
__global__ void k_pool(const float* __restrict__ mm, float* __restrict__ out)
{ // mm [8,1,256,256] -> out [8,1,8,8]
  int t = blockIdx.x * blockDim.x + threadIdx.x;
  if (t >= 512) return;
  int b = t >> 6, r = t & 63, i = r >> 3, j = r & 7;
  const float* p = mm + b * 65536;
  float s = 0.f;
  for (int u = 0; u < 32; ++u)
    for (int v = 0; v < 32; ++v)
      s += p[(i * 32 + u) * 256 + (j * 32 + v)];
  out[t] = s * (1.f / 1024.f);
}

// ---------------------------------------------------------------------------
extern "C" void kernel_launch(void* const* d_in, const int* in_sizes, int n_in,
                              void* d_out, int out_size, void* d_ws, size_t ws_size,
                              hipStream_t stream)
{
  (void)in_sizes; (void)n_in; (void)out_size; (void)ws_size;
  const float** in = (const float**)d_in;
  // Leaf order: dict order {x, embedding, params}; params flattened as a JAX
  // pytree (dict keys sorted alphabetically at each level).
  const float* x       = in[0];
  const float* emb     = in[1];
  const float* bc_b    = in[2];  const float* bc_dw  = in[3];
  const float* bc_pw   = in[4];  const float* bc_g   = in[5];
  const float* ln_b    = in[6];  const float* ln_g   = in[7];
  const float* ssm_A   = in[8];  const float* ssm_Bm = in[9];
  const float* ssm_C   = in[10]; const float* ssm_D  = in[11];
  const float* Wdt     = in[12]; const float* Wout   = in[13];
  const float* Wskip   = in[14]; const float* Wx     = in[15];
  const float* bdt     = in[16]; const float* bout   = in[17];
  const float* bskip   = in[18]; const float* bx     = in[19];
  const float* u1_b    = in[20]; const float* u1_dw  = in[21];  // dec1 (up)
  const float* u1_pw   = in[22]; const float* u1_g   = in[23];
  const float* c1_b    = in[24]; const float* c1_dw  = in[25];  // dec1_conv
  const float* c1_pw   = in[26]; const float* c1_g   = in[27];
  const float* u2_b    = in[28]; const float* u2_dw  = in[29];  // dec2 (up)
  const float* u2_pw   = in[30]; const float* u2_g   = in[31];
  const float* c2_b    = in[32]; const float* c2_dw  = in[33];  // dec2_conv
  const float* c2_pw   = in[34]; const float* c2_g   = in[35];
  const float* eca_k   = in[36];
  const float* en1_b   = in[37]; const float* en1_dw = in[38];
  const float* en1_pw  = in[39]; const float* en1_g  = in[40];
  const float* en2_b   = in[41]; const float* en2_dw = in[42];
  const float* en2_pw  = in[43]; const float* en2_g  = in[44];
  const float* f_b2    = in[45]; const float* f_dw   = in[46];
  const float* f_pw    = in[47]; const float* f_w2   = in[48];
  const float* ic_b    = in[49]; const float* ic_dw  = in[50];
  const float* ic_pw   = in[51]; const float* ic_g   = in[52];
  const float* mean_bb = in[53]; const float* mean_dw = in[54];
  const float* mean_pw = in[55]; const float* mean_w  = in[56];
  const float* oc_dw   = in[57]; const float* oc_pw   = in[58];

  const int HW0 = 65536, HW1 = 16384, HW2 = 4096, T = 4096;
  const long NB16_0 = 8L * 16 * HW0, NB32_0 = 8L * 32 * HW0;
  const long NB32_1 = 8L * 32 * HW1;
  const long NB64_2 = 8L * 64 * HW2, NB16_2 = 8L * 16 * HW2;

  // workspace plan (~355 MB of f32), explicit slot reuse across stages
  float* ws  = (float*)d_ws;
  float* e0   = ws;                    // [8,16,HW0]
  float* e0m  = e0  + NB16_0;          // persistent until dec1 concat
  float* e1b  = e0m + NB16_0;          // [8,32,HW1] persistent until dec2 concat
  float* tA   = e1b + NB32_1;          // [8,32,HW0] big temp
  float* tB   = tA  + NB32_0;          // [8,32,HW0] big temp
  float* tC   = tB  + NB32_0;          // [8,16,HW0] temp
  float* tD   = tC  + NB16_0;          // [8,16,HW0] temp
  float* bk   = tD  + NB16_0;          // bottleneck pool
  float* be2   = bk + 0 * NB64_2;
  float* dwb   = bk + 1 * NB64_2;
  float* bb    = bk + 2 * NB64_2;
  float* lnq   = bk + 3 * NB64_2;
  float* skipb = bk + 4 * NB64_2;
  float* y1b   = bk + 5 * NB64_2;
  float* so6   = bk + 6 * NB64_2;
  float* dtb   = bk + 7 * NB64_2;
  float* xpb   = dtb + NB16_2;
  float* hsf   = xpb + NB16_2;         // scan writes dir0 here, dir1 at +NB16_2
  float* hsav  = hsf + 2 * NB16_2;
  float* sm    = hsav + NB16_2;
  float* femb  = sm;                   // 128
  float* gap   = sm + 128;             // 512
  float* gate  = sm + 640;             // 512

  auto blk1 = [](long n) { return dim3((unsigned)((n + 255) / 256)); };
  auto dw = [&](const float* X, const float* Wk, float* Y, int C, int H, int Wd, int s) {
    long total = 8L * C * (H / s) * (Wd / s);
    k_dw3x3<<<blk1(total), 256, 0, stream>>>(X, Wk, Y, C, H, Wd, s);
  };
  auto gn = [&](float* X, const float* g, const float* b, int C, int HW) {
    k_gn_silu<<<dim3(64), 256, 0, stream>>>(X, g, b, C, HW, 8);
  };

  // ---------------- in_conv: DSConv(3->16) + GN + SiLU
  dw(x, ic_dw, tC, 3, 256, 256, 1);
  launch_pw<3, 3, 16, ACT_NONE, false, false>(tC, ic_pw, nullptr, nullptr, e0, HW0, stream);
  gn(e0, ic_g, ic_b, 16, HW0);

  // ---------------- FiLM (concat dissolved into per-(b,o) WMMA epilogue bias)
  dw(e0, f_dw, tC, 16, 256, 256, 1);                       // dw on x-channels 0..15
  k_femb<<<dim3(1), 128, 0, stream>>>(emb, f_pw, f_dw, femb);
  launch_pw<16, 48, 16, ACT_SILU, false, true>(tC, f_pw, nullptr, femb, tD, HW0, stream);
  launch_pw<16, 16, 32, ACT_NONE, true, false>(tD, f_w2, f_b2, nullptr, tA, HW0, stream);
  k_film_apply<<<blk1(NB16_0), 256, 0, stream>>>(e0, tA, e0m, HW0);

  // ---------------- enc1 (stride 2), enc2 (stride 2)
  dw(e0m, en1_dw, tC, 16, 256, 256, 2);
  launch_pw<16, 16, 32, ACT_NONE, false, false>(tC, en1_pw, nullptr, nullptr, e1b, HW1, stream);
  gn(e1b, en1_g, en1_b, 32, HW1);
  dw(e1b, en2_dw, tC, 32, 128, 128, 2);
  launch_pw<32, 32, 64, ACT_NONE, false, false>(tC, en2_pw, nullptr, nullptr, be2, HW2, stream);
  gn(be2, en2_g, en2_b, 64, HW2);

  // ---------------- bottleneck conv
  dw(be2, bc_dw, dwb, 64, 64, 64, 1);
  launch_pw<64, 64, 64, ACT_NONE, false, false>(dwb, bc_pw, nullptr, nullptr, bb, HW2, stream);
  gn(bb, bc_g, bc_b, 64, HW2);

  // ---------------- SSM block (bidirectional collapsed: avg hidden states)
  k_ln_chan<<<blk1(8L * HW2), 256, 0, stream>>>(bb, ln_g, ln_b, lnq, 64, HW2);
  launch_pw<64, 64, 16, ACT_SP,   true, false>(lnq, Wdt,   bdt,   nullptr, dtb,   HW2, stream);
  launch_pw<64, 64, 16, ACT_NONE, true, false>(lnq, Wx,    bx,    nullptr, xpb,   HW2, stream);
  launch_pw<64, 64, 64, ACT_NONE, true, false>(lnq, Wskip, bskip, nullptr, skipb, HW2, stream);
  k_ssm_scan<<<dim3(8, 2), 256, 0, stream>>>(dtb, xpb, ssm_A, ssm_Bm, hsf, T);
  k_avg<<<blk1(NB16_2), 256, 0, stream>>>(hsav, hsf, hsf + NB16_2, (int)NB16_2);
  launch_pw<16, 16, 64, ACT_NONE, false, false>(hsav, ssm_C, nullptr, nullptr, y1b, HW2, stream);
  k_yfix<<<blk1(NB64_2), 256, 0, stream>>>(y1b, lnq, ssm_D, 64, HW2, (int)NB64_2);
  launch_pw<64, 64, 64, ACT_NONE, true, false>(y1b, Wout, bout, nullptr, so6, HW2, stream);
  k_add<<<blk1(NB64_2), 256, 0, stream>>>(so6, skipb, (int)NB64_2);

  // ---------------- ECA gate
  k_gap<<<dim3(512), 256, 0, stream>>>(so6, gap, HW2);
  k_eca_gate<<<dim3(2), 256, 0, stream>>>(gap, eca_k, gate);
  k_scale_chan<<<blk1(NB64_2), 256, 0, stream>>>(so6, gate, HW2, (int)NB64_2);

  // ---------------- dec2: up(64->128, ps->32) + concat(e1) + conv(64->32)
  dw(so6, u2_dw, dwb, 64, 64, 64, 1);
  launch_pw<64, 64, 128, ACT_NONE, false, false>(dwb, u2_pw, nullptr, nullptr, tC, HW2, stream);
  k_pixshuf<<<blk1(8L * 32 * HW1), 256, 0, stream>>>(tC, tD, 32, 64, 64);
  gn(tD, u2_g, u2_b, 32, HW1);
  k_copy_chan<<<blk1(8L * 32 * HW1), 256, 0, stream>>>(tA, tD, 64, 0, 32, HW1);
  k_copy_chan<<<blk1(8L * 32 * HW1), 256, 0, stream>>>(tA, e1b, 64, 32, 32, HW1);
  dw(tA, c2_dw, tB, 64, 128, 128, 1);
  launch_pw<64, 64, 32, ACT_NONE, false, false>(tB, c2_pw, nullptr, nullptr, tC, HW1, stream);
  gn(tC, c2_g, c2_b, 32, HW1);

  // ---------------- dec1: up(32->64, ps->16) + concat(e0m) + conv(32->16)
  dw(tC, u1_dw, tD, 32, 128, 128, 1);
  launch_pw<32, 32, 64, ACT_NONE, false, false>(tD, u1_pw, nullptr, nullptr, tB, HW1, stream);
  k_pixshuf<<<blk1(NB16_0), 256, 0, stream>>>(tB, tD, 16, 128, 128);
  gn(tD, u1_g, u1_b, 16, HW0);
  k_copy_chan<<<blk1(NB16_0), 256, 0, stream>>>(tA, tD, 32, 0, 16, HW0);
  k_copy_chan<<<blk1(NB16_0), 256, 0, stream>>>(tA, e0m, 32, 16, 16, HW0);
  dw(tA, c1_dw, tB, 32, 256, 256, 1);
  launch_pw<32, 32, 16, ACT_NONE, false, false>(tB, c1_pw, nullptr, nullptr, tC, HW0, stream);
  gn(tC, c1_g, c1_b, 16, HW0);                              // tC = d1

  // ---------------- heads: fin = DSConv(16->3); pooled exposure map
  float* outp = (float*)d_out;
  dw(tC, oc_dw, tD, 16, 256, 256, 1);
  launch_pw<16, 16, 3, ACT_NONE, false, false>(tD, oc_pw, nullptr, nullptr, outp, HW0, stream);
  dw(tC, mean_dw, tD, 16, 256, 256, 1);
  launch_pw<16, 16, 8, ACT_SILU, false, false>(tD, mean_pw, nullptr, nullptr, tB, HW0, stream);
  launch_pw<8, 8, 1, ACT_SIG, true, false>(tB, mean_w, mean_bb, nullptr, tA, HW0, stream);
  k_pool<<<dim3(2), 256, 0, stream>>>(tA, outp + 8L * 3 * HW0);
}
